// GraphLoss_39058432590195
// MI455X (gfx1250) — matrix operand order repaired and checked by reading, not compiled, not run on hardware
//
#include <hip/hip_runtime.h>
#include <hip/hip_bf16.h>
#include <math.h>

typedef __attribute__((ext_vector_type(16))) _Float16 v16h;
typedef __attribute__((ext_vector_type(8)))  _Float16 v8h;
typedef __attribute__((ext_vector_type(8)))  float    v8f;

#define NNODE 8192
#define FDIM  256
#define NEDGE 262144
#define ALPHA_C   0.9f
#define EPS_C     0.01f
#define LAMBDA_C  1.0f
#define THRESH_C  0.8f
#define COSEPS_C  1e-8f

// ---------------- WMMA helpers (CDNA5 wave32, 16x16x32 f16 -> f32) ----------

__device__ __forceinline__ v8f wmma_f16(v16h a, v16h b, v8f c) {
    // 8 args: (neg_a, A, neg_b, B, c_mod, C, reuse_a, reuse_b)
    return __builtin_amdgcn_wmma_f32_16x16x32_f16(false, a, false, b,
                                                  (short)0, c, false, false);
}

// A fragment 16x32 (MxK), K contiguous in memory. p -> element (m=0,k=0),
// rs = stride between M rows (elements). Per ISA 7.12.2 the per-lane data is
// two contiguous 8-half chunks: k = kb..kb+7 and kb+16..kb+23 (kb = 0|8).
__device__ __forceinline__ v16h load_a_k1(const _Float16* p, long rs, int lane) {
    int m  = lane & 15;
    int kb = (lane < 16) ? 0 : 8;
    const _Float16* q = p + (long)m * rs + kb;
    v8h lo = *(const v8h*)(q);        // global_load_b128
    v8h hi = *(const v8h*)(q + 16);   // global_load_b128
    return __builtin_shufflevector(lo, hi, 0, 1, 2, 3, 4, 5, 6, 7,
                                   8, 9, 10, 11, 12, 13, 14, 15);
}

// B fragment 32x16 (KxN), K contiguous. p -> element (k=0,n=0), ns = stride
// between N columns (elements). Per lane: two contiguous 8-half chunks
// k = kb..kb+7 and kb+16..kb+23 (kb = 0|8).
__device__ __forceinline__ v16h load_b_k1(const _Float16* p, long ns, int lane) {
    int n  = lane & 15;
    int kb = (lane < 16) ? 0 : 8;
    const _Float16* q = p + (long)n * ns + kb;
    v8h lo = *(const v8h*)(q);
    v8h hi = *(const v8h*)(q + 16);
    return __builtin_shufflevector(lo, hi, 0, 1, 2, 3, 4, 5, 6, 7,
                                   8, 9, 10, 11, 12, 13, 14, 15);
}

__device__ __forceinline__ float wave_reduce_add(float v) {
#pragma unroll
    for (int o = 16; o > 0; o >>= 1) v += __shfl_down(v, o, 32);
    return v;
}

// ---------------- Utility kernels ----------------

__global__ void k_zero(unsigned* p, unsigned n) {
    unsigned i = blockIdx.x * blockDim.x + threadIdx.x;
    unsigned s = gridDim.x * blockDim.x;
    for (; i < n; i += s) p[i] = 0u;
}

__global__ void k_colsum(const float* __restrict__ z, float* __restrict__ mean) {
    int f  = threadIdx.x;
    int r0 = blockIdx.x * 128;
    float s = 0.f;
    for (int r = 0; r < 128; ++r) s += z[(size_t)(r0 + r) * FDIM + f];
    atomicAdd(&mean[f], s);
}

__global__ void k_scale_mean(float* mean) {
    mean[threadIdx.x] *= (1.0f / (float)NNODE);
}

__global__ void k_weights(const int* __restrict__ k2u, const int* __restrict__ u2k,
                          float* __restrict__ w) {
    int n = blockIdx.x * blockDim.x + threadIdx.x;
    if (n < NNODE) {
        float a = powf(ALPHA_C, (float)k2u[n]);
        float b = powf(ALPHA_C, (float)u2k[n]);
        w[n] = a + 1.0f - b;
    }
}

// zm = z - mean; zmhT[f*N+n] = f16(zm) (transposed, K-contiguous for cov);
// wh[n*F+f] = f16(w[n]*zm)
__global__ void k_prep(const float* __restrict__ z, const float* __restrict__ mean,
                       const float* __restrict__ w,
                       _Float16* __restrict__ zmhT, _Float16* __restrict__ wh) {
    unsigned i = blockIdx.x * blockDim.x + threadIdx.x;
    unsigned s = gridDim.x * blockDim.x;
    for (; i < (unsigned)NNODE * FDIM; i += s) {
        unsigned n = i / FDIM, f = i % FDIM;
        float zm = z[i] - mean[f];
        zmhT[(size_t)f * NNODE + n] = (_Float16)zm;
        wh[i] = (_Float16)(w[n] * zm);
    }
}

// ---------------- WMMA kernels ----------------

// cov[fi,fj] = sum_n zm[n,fi]*zm[n,fj]: both operands from zmhT (K=node contiguous)
__global__ void k_cov(const _Float16* __restrict__ zmhT, float* __restrict__ cov) {
    int lane = threadIdx.x;
    int fi0 = blockIdx.y * 16, fj0 = blockIdx.x * 16;
    v8f acc = {};
    for (int k0 = 0; k0 < NNODE; k0 += 32) {
        v16h a = load_a_k1(zmhT + (size_t)fi0 * NNODE + k0, NNODE, lane);
        v16h b = load_b_k1(zmhT + (size_t)fj0 * NNODE + k0, NNODE, lane);
        acc = wmma_f16(a, b, acc);
    }
    int col = lane & 15, rb = (lane >> 4) * 8;
#pragma unroll
    for (int v = 0; v < 8; ++v)
        cov[(size_t)(fi0 + rb + v) * FDIM + fj0 + col] = acc[v];
}

__global__ void k_cor(const float* __restrict__ cov, _Float16* __restrict__ corh) {
    unsigned i = blockIdx.x * blockDim.x + threadIdx.x;
    if (i < (unsigned)FDIM * FDIM) {
        unsigned fi = i / FDIM, fj = i % FDIM;
        float d = sqrtf(cov[(size_t)fi * FDIM + fi]) * sqrtf(cov[(size_t)fj * FDIM + fj]);
        float v = cov[i] / d;
        if (v != v) v = 0.f;
        v = fminf(1.f, fmaxf(-1.f, v));
        if (fi == fj) v = 0.f;
        corh[i] = (_Float16)v;
    }
}

// z2 = z + EPS * (W @ cor). B uses cor symmetry: cor[k,n] = cor[n,k].
__global__ void k_z2(const _Float16* __restrict__ wh, const _Float16* __restrict__ corh,
                     const float* __restrict__ z,
                     float* __restrict__ z2, _Float16* __restrict__ z2h) {
    int lane = threadIdx.x;
    int ni0 = blockIdx.y * 16;
    int fj0 = blockIdx.x * 16;
    v8f acc = {};
#pragma unroll
    for (int k0 = 0; k0 < FDIM; k0 += 32) {
        v16h a = load_a_k1(wh + (size_t)ni0 * FDIM + k0, FDIM, lane);
        v16h b = load_b_k1(corh + (size_t)fj0 * FDIM + k0, FDIM, lane);
        acc = wmma_f16(a, b, acc);
    }
    int col = lane & 15, rb = (lane >> 4) * 8;
#pragma unroll
    for (int v = 0; v < 8; ++v) {
        size_t gi = (size_t)(ni0 + rb + v) * FDIM + fj0 + col;
        float val = z[gi] + EPS_C * acc[v];
        z2[gi]  = val;
        z2h[gi] = (_Float16)val;
    }
}

// ---------------- graph-side kernels ----------------

__global__ void k_norm(const float* __restrict__ z2, float* __restrict__ z2n) {
    __shared__ float red[256];
    int n = blockIdx.x, t = threadIdx.x;
    float v = z2[(size_t)n * FDIM + t];
    red[t] = v * v;
    __syncthreads();
    for (int s = 128; s > 0; s >>= 1) {
        if (t < s) red[t] += red[t + s];
        __syncthreads();
    }
    if (t == 0) z2n[n] = fmaxf(sqrtf(red[0]), COSEPS_C);
}

__global__ void k_adj(const int* __restrict__ ei, unsigned* __restrict__ adj) {
    int e = blockIdx.x * blockDim.x + threadIdx.x;
    if (e < NEDGE) {
        unsigned i = (unsigned)ei[e];
        unsigned j = (unsigned)ei[NEDGE + e];
        unsigned bi = i * NNODE + j;
        unsigned bj = j * NNODE + i;
        atomicOr(&adj[bi >> 5], 1u << (bi & 31));
        atomicOr(&adj[bj >> 5], 1u << (bj & 31));
    }
}

// one wave per edge: cosine sim via float4 loads, scatter-add by source node
__global__ void k_edge(const int* __restrict__ ei, const float* __restrict__ z2,
                       const float* __restrict__ z2n,
                       float* __restrict__ scores, float* __restrict__ deg) {
    int wid = threadIdx.x >> 5, lane = threadIdx.x & 31;
    int e = blockIdx.x * 8 + wid;
    unsigned i = (unsigned)ei[e];
    unsigned j = (unsigned)ei[NEDGE + e];
    const float4* zi = (const float4*)(z2 + (size_t)i * FDIM);
    const float4* zj = (const float4*)(z2 + (size_t)j * FDIM);
    float s = 0.f;
#pragma unroll
    for (int c = 0; c < 2; ++c) {
        float4 a = zi[lane + 32 * c];
        float4 b = zj[lane + 32 * c];
        s += a.x * b.x + a.y * b.y + a.z * b.z + a.w * b.w;
    }
    s = wave_reduce_add(s);
    if (lane == 0) {
        float sim = s / (z2n[i] * z2n[j]);
        atomicAdd(&scores[i], sim);
        atomicAdd(&deg[i], 1.0f);
    }
}

// gram: sim = z2 @ z2^T with fused adjacency-mask threshold reduction.
// Block = 8 waves, owns 8 column tiles = 128 contiguous z2h rows (64 KB),
// staged to LDS with global_load_async_to_lds_b128 (ASYNCcnt). Each wave:
// per row tile, load 8 A fragments once (2x b128 each), then reuse them
// across all 8 column tiles with B fragments streamed from LDS
// (2x ds_load_b128 per WMMA). 8x less L2 traffic than B-from-global.
__global__ void k_gram(const _Float16* __restrict__ z2h, const unsigned* __restrict__ adj,
                       float* __restrict__ acc_pen) {
    __shared__ __align__(16) _Float16 Bs[128 * FDIM];   // 64 KB: 8 col tiles

    int lane = threadIdx.x;
    int wv   = threadIdx.y;
    int tid  = wv * 32 + lane;
    int tjBase = blockIdx.x * 128;                      // first column of block

    // ---- async stage: 128 rows of z2h (contiguous 64 KB) -> LDS ----
    {
        unsigned ldsbase = (unsigned)(uintptr_t)(void*)Bs;  // low 32 bits = LDS offset
        const _Float16* gsrc = z2h + (size_t)tjBase * FDIM;
        for (int c = tid; c < 4096; c += 256) {             // 4096 x 16B chunks
            unsigned laddr = ldsbase + (unsigned)c * 16u;
            const _Float16* g = gsrc + (size_t)c * 8;
            asm volatile("global_load_async_to_lds_b128 %0, %1, off"
                         :: "v"(laddr), "v"(g) : "memory");
        }
        asm volatile("s_wait_asynccnt 0x0" ::: "memory");
    }
    __syncthreads();

    // ---- sweep 8 row tiles per wave, 8 column tiles each ----
    float local = 0.f;
    int col = lane & 15, rb = (lane >> 4) * 8;
    for (int rt = 0; rt < 8; ++rt) {
        int ti0 = ((blockIdx.y * 8 + wv) * 8 + rt) * 16;
        const _Float16* pA = z2h + (size_t)ti0 * FDIM;
        v16h af[8];
#pragma unroll
        for (int k0 = 0; k0 < 8; ++k0)
            af[k0] = load_a_k1(pA + k0 * 32, FDIM, lane);

        for (int ct = 0; ct < 8; ++ct) {
            v8f acc = {};
#pragma unroll
            for (int k0 = 0; k0 < 8; ++k0) {
                // B[k,n] = Bs[(ct*16+n)*F + k0*32 + k]   (ds_load_b128 x2)
                v16h b = load_b_k1(Bs + (size_t)ct * 16 * FDIM + k0 * 32, FDIM, lane);
                acc = wmma_f16(af[k0], b, acc);
            }
            int tj0 = tjBase + ct * 16;
#pragma unroll
            for (int v = 0; v < 8; ++v) {
                unsigned gi  = (unsigned)(ti0 + rb + v);
                unsigned gj  = (unsigned)(tj0 + col);
                unsigned bit = gi * NNODE + gj;
                bool connected = (adj[bit >> 5] >> (bit & 31)) & 1u;
                float s = acc[v];
                if (!connected && s > THRESH_C) local += s;
            }
        }
    }
    local = wave_reduce_add(local);
    if (lane == 0) atomicAdd(acc_pen, local);
}

__global__ void k_final(const float* __restrict__ scores, const float* __restrict__ deg,
                        const float* __restrict__ acc_pen, float* __restrict__ out) {
    __shared__ float red[256];
    int t = threadIdx.x;
    float s = 0.f;
    for (int n = t; n < NNODE; n += 256) {
        float d = deg[n];
        if (d == 0.f) d = 1.f;
        s += scores[n] / d;
    }
    red[t] = s;
    __syncthreads();
    for (int st = 128; st > 0; st >>= 1) {
        if (t < st) red[t] += red[t + st];
        __syncthreads();
    }
    if (t == 0) {
        float homophily = -(red[0] / (float)NNODE);
        float penalty = acc_pen[0] / ((float)NNODE * (float)NNODE);
        out[0] = homophily + LAMBDA_C * penalty;
    }
}

// ---------------- launcher ----------------

extern "C" void kernel_launch(void* const* d_in, const int* in_sizes, int n_in,
                              void* d_out, int out_size, void* d_ws, size_t ws_size,
                              hipStream_t stream) {
    const float* z   = (const float*)d_in[0];
    // d_in[1] = x_hat (unused by the reference)
    const int*   ei  = (const int*)d_in[2];
    const int*   k2u = (const int*)d_in[3];
    const int*   u2k = (const int*)d_in[4];
    // d_in[5] = epoch (unused)
    float* out = (float*)d_out;

    char* ws = (char*)d_ws;
    float*     mean   = (float*)(ws + 0);               //  1 KB
    float*     wvec   = (float*)(ws + 1024);            // 32 KB
    float*     scores = (float*)(ws + 33792);           // 32 KB
    float*     deg    = (float*)(ws + 66560);           // 32 KB
    float*     accp   = (float*)(ws + 99328);           // penalty accumulator
    float*     z2n    = (float*)(ws + 99584);           // 32 KB
    float*     cov    = (float*)(ws + 132352);          // 256 KB
    _Float16*  corh   = (_Float16*)(ws + 394496);       // 128 KB
    _Float16*  zmhT   = (_Float16*)(ws + ((size_t)1  << 20));  // 4 MB [F][N]
    _Float16*  wh     = (_Float16*)(ws + ((size_t)5  << 20));  // 4 MB [N][F]
    float*     z2     = (float*)(ws + ((size_t)9  << 20));     // 8 MB
    _Float16*  z2h    = (_Float16*)(ws + ((size_t)17 << 20));  // 4 MB
    unsigned*  adj    = (unsigned*)(ws + ((size_t)21 << 20));  // 8 MB bitmask

    k_zero<<<4, 256, 0, stream>>>((unsigned*)mean, 256);
    k_zero<<<64, 256, 0, stream>>>((unsigned*)scores, (99584u - 33792u) / 4u);
    k_zero<<<2048, 256, 0, stream>>>(adj, (unsigned)NNODE * (NNODE / 32));

    k_colsum<<<64, 256, 0, stream>>>(z, mean);
    k_scale_mean<<<1, 256, 0, stream>>>(mean);
    k_weights<<<NNODE / 256, 256, 0, stream>>>(k2u, u2k, wvec);
    k_prep<<<2048, 256, 0, stream>>>(z, mean, wvec, zmhT, wh);

    k_cov<<<dim3(FDIM / 16, FDIM / 16), 32, 0, stream>>>(zmhT, cov);
    k_cor<<<(FDIM * FDIM) / 256, 256, 0, stream>>>(cov, corh);
    k_z2<<<dim3(FDIM / 16, NNODE / 16), 32, 0, stream>>>(wh, corh, z, z2, z2h);

    k_norm<<<NNODE, 256, 0, stream>>>(z2, z2n);
    k_adj<<<NEDGE / 256, 256, 0, stream>>>(ei, adj);
    k_edge<<<NEDGE / 8, 256, 0, stream>>>(ei, z2, z2n, scores, deg);

    // dominant phase: 8192x8192 gram. grid: 64 column groups x 8 row groups;
    // block = 8 waves, 64 KB LDS of column data, A reused across 8 col tiles.
    k_gram<<<dim3(NNODE / 128, 8), dim3(32, 8), 0, stream>>>(z2h, adj, accp);

    k_final<<<1, 256, 0, stream>>>(scores, deg, accp, out);
}
